// RNN_scan_cell_23854248362164
// MI455X (gfx1250) — matrix operand order repaired and checked by reading, not compiled, not run on hardware
//
#include <hip/hip_runtime.h>

typedef __attribute__((ext_vector_type(16))) _Float16 v16h;
typedef __attribute__((ext_vector_type(8)))  _Float16 v8h;
typedef __attribute__((ext_vector_type(8)))  float    v8f;

#define T_STEPS 2048
#define B_SZ    256
#define I_SZ    50
#define H_SZ    200
#define HS      264   // LDS row stride in halves (528B = 4 dwords mod 64 banks -> conflict-free)
#define XOFF    200   // combined K: [0,200)=h, [200,250)=x, 250=const-1 (bias), [251,256)=0
#define KBIAS   250

union V16 { v16h v; v8h h[2]; };

__device__ __forceinline__ float fast_tanh(float x) {
#if __has_builtin(__builtin_amdgcn_tanhf)
    return __builtin_amdgcn_tanhf(x);            // V_TANH_F32 (TRANS), branchless
#else
    float e = __builtin_amdgcn_exp2f(x * 2.88539008177792681f);
    return 1.0f - 2.0f * __builtin_amdgcn_rcpf(e + 1.0f);
#endif
}

__device__ __forceinline__ void sched_fence() {
#if __has_builtin(__builtin_amdgcn_sched_barrier)
    __builtin_amdgcn_sched_barrier(0);           // nothing may be moved across
#endif
}

__global__ __launch_bounds__(512, 1)
void rnn_scan_wmma_kernel(const float* __restrict__ init,
                          const float* __restrict__ xs,
                          const float* __restrict__ W_ih,
                          const float* __restrict__ W_hh,
                          const float* __restrict__ b_ih,
                          const float* __restrict__ b_hh,
                          float* __restrict__ out)
{
    // ping-pong combined operand buffer: 16 batch rows x K (h | x | 1 | pad)
    __shared__ __align__(16) _Float16 sbuf[2][16][HS];

    const int tid  = threadIdx.x;
    const int bg   = blockIdx.x;          // batch rows [bg*16, bg*16+16)
    const int lane = tid & 31;
    const int wave = tid >> 5;            // 16 waves, one N-tile each (13..15 zero tiles)
    const int lo   = lane & 15;
    const bool hi  = lane >= 16;

    // ------- one-time: combined weight B-fragments (W_hh | W_ih | bias) into VGPRs -------
    // B-frag (32x16 f16): lane<16 -> n=lane&15, K=kb..kb+15 ; lane>=16 -> K=kb+16..kb+31
    v16h wc[8];
    const int n = wave * 16 + lo;        // output column this lane owns
    const bool nvalid = n < H_SZ;

    #pragma unroll
    for (int kf = 0; kf < 8; ++kf) {
        V16 u;
        const int kb = kf * 32 + (hi ? 16 : 0);
        #pragma unroll
        for (int j = 0; j < 16; ++j) {
            const int kk = kb + j;
            float w = 0.0f;
            if (nvalid) {
                if (kk < H_SZ)              w = W_hh[n * H_SZ + kk];
                else if (kk < XOFF + I_SZ)  w = W_ih[n * I_SZ + (kk - XOFF)];
                else if (kk == KBIAS)       w = b_ih[n] + b_hh[n];   // bias via const-1 col
            }
            u.v[j] = (_Float16)w;
        }
        wc[kf] = u.v;
    }

    // ------- one-time: stage h0 into sbuf[0] -------
    for (int s = tid; s < 16 * H_SZ; s += 512) {
        const int m = s / H_SZ, k = s % H_SZ;
        sbuf[0][m][k] = (_Float16)init[(bg * 16 + m) * H_SZ + k];
    }

    // x slot geometry: 1024 slots = 16 rows x 64 cols (cols 200..263); 2 slots/thread.
    // Addresses clamped in-bounds -> unconditional loads; masking at the staging store.
    int   xm[2], xc[2];
    long  xgo[2];
    bool  xdat[2];
    float xfill[2];
    #pragma unroll
    for (int j = 0; j < 2; ++j) {
        const int s = tid + 512 * j;
        xm[j] = s >> 6;
        xc[j] = s & 63;
        xdat[j]  = xc[j] < I_SZ;
        xfill[j] = (xc[j] == I_SZ) ? 1.0f : 0.0f;    // col 250 = constant 1 (bias column)
        xgo[j]   = (long)(bg * 16 + xm[j]) * I_SZ + (xdat[j] ? xc[j] : 0);
    }

    const long stepXS  = (long)B_SZ * I_SZ;
    const long stepOut = (long)B_SZ * H_SZ;
    const int  mb      = bg * 16 + (hi ? 8 : 0);

    // stage x(0); preload x(1) into registers
    float xr[2];
    #pragma unroll
    for (int j = 0; j < 2; ++j) {
        const float v = xs[xgo[j]];
        sbuf[0][xm[j]][XOFF + xc[j]] = (_Float16)(xdat[j] ? v : xfill[j]);
        xr[j] = xs[stepXS + xgo[j]];                 // T_STEPS > 1
    }
    __syncthreads();

    // ------- the scan -------
    for (int t = 0; t < T_STEPS; ++t) {
        const int cur = t & 1, nxt = cur ^ 1;

        // issue loads for x(t+2) (clamped, unconditional); prefetch x(t+4)
        const int tld = (t + 2 < T_STEPS) ? (t + 2) : (T_STEPS - 1);
        const int tpf = (t + 4 < T_STEPS) ? (t + 4) : (T_STEPS - 1);
        float xn[2];
        #pragma unroll
        for (int j = 0; j < 2; ++j) {
            xn[j] = xs[(long)tld * stepXS + xgo[j]];
            __builtin_prefetch(&xs[(long)tpf * stepXS + xgo[j]], 0, 1);
        }

        // --- burst-load ALL 8 A-fragments; sched_barrier pins the loads above the
        //     WMMA chain so all 8 fragments are simultaneously live (distinct regs)
        //     and the dscnt waits decay instead of stalling per-WMMA ---
        const _Float16* base = &sbuf[cur][0][0];
        const int arow = lo * HS + (hi ? 16 : 0);

        V16 af[8];
        #pragma unroll
        for (int kf = 0; kf < 8; ++kf) {
            const _Float16* p = base + arow + kf * 32;
            af[kf].h[0] = *(const v8h*)(p);
            af[kf].h[1] = *(const v8h*)(p + 8);
        }

        sched_fence();   // loads stay above, WMMAs stay below

        v8f c = {};   // bias comes through the const-1 column; SRC2 starts as inline 0
        #pragma unroll
        for (int kf = 0; kf < 8; ++kf)
            c = __builtin_amdgcn_wmma_f32_16x16x32_f16(
                    false, af[kf].v, false, wc[kf], (short)0, c, false, false);

        // --- tanh, publish h_new (f16 -> LDS), stream outs (f32 -> HBM, NT) ---
        const long orow = stepOut + (long)t * stepOut;   // outs after carry [B,H]
        #pragma unroll
        for (int r = 0; r < 8; ++r) {
            const float hv = fast_tanh(c[r]);
            const int m = r + (hi ? 8 : 0);
            if (nvalid) {
                sbuf[nxt][m][n] = (_Float16)hv;          // only n<200: x region untouched
                __builtin_nontemporal_store(hv, &out[orow + (long)(mb + r) * H_SZ + n]);
                if (t == T_STEPS - 1)                    // carry = final h (uniform branch)
                    out[(long)(mb + r) * H_SZ + n] = hv;
            }
        }

        // stage x(t+1) (loaded last iteration) — unconditional stores, masked values
        #pragma unroll
        for (int j = 0; j < 2; ++j)
            sbuf[nxt][xm[j]][XOFF + xc[j]] = (_Float16)(xdat[j] ? xr[j] : xfill[j]);
        #pragma unroll
        for (int j = 0; j < 2; ++j) xr[j] = xn[j];

        __syncthreads();   // ping-pong -> one barrier per step
    }
}

extern "C" void kernel_launch(void* const* d_in, const int* in_sizes, int n_in,
                              void* d_out, int out_size, void* d_ws, size_t ws_size,
                              hipStream_t stream) {
    (void)in_sizes; (void)n_in; (void)out_size; (void)d_ws; (void)ws_size;
    const float* init = (const float*)d_in[0];
    const float* xs   = (const float*)d_in[1];
    const float* W_ih = (const float*)d_in[2];
    const float* W_hh = (const float*)d_in[3];
    const float* b_ih = (const float*)d_in[4];
    const float* b_hh = (const float*)d_in[5];
    float* out = (float*)d_out;

    // 16 persistent workgroups (one per 16-row batch slice), 16 waves each (1 N-tile/wave).
    rnn_scan_wmma_kernel<<<dim3(B_SZ / 16), dim3(512), 0, stream>>>(
        init, xs, W_ih, W_hh, b_ih, b_hh, out);
}
